// FullAttention_33062658245138
// MI455X (gfx1250) — matrix operand order repaired and checked by reading, not compiled, not run on hardware
//
#include <hip/hip_runtime.h>

typedef __attribute__((ext_vector_type(16))) _Float16 v16h;
typedef __attribute__((ext_vector_type(8)))  _Float16 v8h;
typedef __attribute__((ext_vector_type(8)))  float    v8f;

#define B_     4
#define L_     2048
#define H_     8
#define E_     64
#define SCHUNK 64
#define BM     64     // query rows per workgroup (4 waves x 16)
#define NT     128    // 4 wave32

union FragH { v16h v; _Float16 h[16]; unsigned u[8]; };
union PairH { _Float16 h[2]; unsigned u; };

// A-fragment k striping (16-bit A 16x32, ISA 7.12.2): lanes<16 hold K {0..7,16..23},
// lanes>=16 hold K {8..15,24..31}, packed 2 per VGPR.
__device__ __forceinline__ int k0A(int j, int hf) {
  return (j < 4) ? (2 * j + 8 * hf) : (16 + 2 * (j - 4) + 8 * hf);
}
// B-fragment k striping (32x16): lanes<16 hold K {0..7,16..23}, lanes>=16 hold K {8..15,24..31}.
__device__ __forceinline__ int k0B(int j, int hf) {
  return (j < 4) ? (2 * j + 8 * hf) : (16 + 2 * (j - 4) + 8 * hf);
}

// 16x32 f16 A-fragment from LDS tile stored row-major [m][ld] (ld in halfs).
__device__ __forceinline__ v16h frag_a_lds(const _Float16* base, int ld, int lm, int hf) {
  FragH f;
  const unsigned* b32 = (const unsigned*)base;
#pragma unroll
  for (int j = 0; j < 8; ++j)
    f.u[j] = b32[(lm * ld + k0A(j, hf)) >> 1];
  return f.v;
}
// 32x16 f16 B-fragment; source stored row-major [n][k] (n = output column).
__device__ __forceinline__ v16h frag_b_lds(const _Float16* base, int ld, int lm, int hf) {
  FragH f;
  const unsigned* b32 = (const unsigned*)base;
#pragma unroll
  for (int j = 0; j < 8; ++j)
    f.u[j] = b32[(lm * ld + k0B(j, hf)) >> 1];
  return f.v;
}
// B-fragment from a global f32 row, pre-scaled and converted to f16.
__device__ __forceinline__ v16h frag_b_global_f32(const float* rowbase, int hf, float s) {
  FragH f;
#pragma unroll
  for (int j = 0; j < 8; ++j) {
    int k = k0B(j, hf);
    f.h[2 * j]     = (_Float16)(rowbase[k]     * s);
    f.h[2 * j + 1] = (_Float16)(rowbase[k + 1] * s);
  }
  return f.v;
}

// attn(Q, K) = softmax(Q K^T / sqrt(E)) @ K, computed transposed in log2 domain:
//   S^T = K_chunk * (Q * scale*log2e)^T  -> WMMA output is already log2-domain
//   V1^T = K_chunk^T * P^T
__global__ __launch_bounds__(NT) void attn_pass(const float* __restrict__ Q,
                                                const float* __restrict__ K,
                                                float* __restrict__ Out) {
  __shared__ __align__(16) _Float16 Krm[SCHUNK * E_];     // [s][e]      8 KB
  __shared__ __align__(16) _Float16 Kt[E_ * SCHUNK];      // [d][s]      8 KB
  __shared__ __align__(16) _Float16 Pw[4][16 * SCHUNK];   // P^T [l][s]  8 KB

  const int tid  = threadIdx.x;
  const int wave = tid >> 5;
  const int lane = tid & 31;
  const int hf   = lane >> 4;
  const int lm   = lane & 15;

  const int bh = blockIdx.y;
  const int b  = bh / H_;
  const int h  = bh % H_;
  const int m0 = blockIdx.x * BM + wave * 16;   // first query row of this wave

  // Q tile as B-fragments with (1/sqrt(E))*log2(e) folded in at conversion time.
  const float K2 = 0.18033688f;
  const float* qrow = Q + (((size_t)b * L_ + (m0 + lm)) * H_ + h) * E_;
  v16h bq0 = frag_b_global_f32(qrow,      hf, K2);   // e = 0..31
  v16h bq1 = frag_b_global_f32(qrow + 32, hf, K2);   // e = 32..63

  float mrun = -3.0e30f, lrun = 0.0f;                // log2-domain running max/sum
  v8f acc[4];
#pragma unroll
  for (int t = 0; t < 4; ++t) { v8f z = {}; acc[t] = z; }

  // loader mapping: each thread handles a 4-row x 8-col f32 patch of the K chunk
  const int sp = tid >> 3;          // 0..15 -> rows 4sp..4sp+3
  const int e0 = (tid & 7) * 8;     // 8-col slice

  for (int s0 = 0; s0 < L_; s0 += SCHUNK) {
    // -------- cooperative K-chunk load (64x64 f32 -> f16, two LDS layouts)
    const float* kr =
        K + (((size_t)b * L_ + (s0 + 4 * sp)) * H_ + h) * E_ + e0;
    float4 kv[4][2];
#pragma unroll
    for (int rr = 0; rr < 4; ++rr) {
      kv[rr][0] = ((const float4*)(kr + (size_t)rr * H_ * E_))[0];
      kv[rr][1] = ((const float4*)(kr + (size_t)rr * H_ * E_))[1];
    }
    if (s0 + SCHUNK < L_)
      __builtin_prefetch(kr + (size_t)SCHUNK * H_ * E_, 0, 0);

    union { v8h v; _Float16 h[8]; } rh[4];
#pragma unroll
    for (int rr = 0; rr < 4; ++rr) {
      rh[rr].h[0] = (_Float16)kv[rr][0].x; rh[rr].h[1] = (_Float16)kv[rr][0].y;
      rh[rr].h[2] = (_Float16)kv[rr][0].z; rh[rr].h[3] = (_Float16)kv[rr][0].w;
      rh[rr].h[4] = (_Float16)kv[rr][1].x; rh[rr].h[5] = (_Float16)kv[rr][1].y;
      rh[rr].h[6] = (_Float16)kv[rr][1].z; rh[rr].h[7] = (_Float16)kv[rr][1].w;
    }

    __syncthreads();   // previous iteration's LDS reads complete
#pragma unroll
    for (int rr = 0; rr < 4; ++rr)
      *(v8h*)&Krm[(4 * sp + rr) * E_ + e0] = rh[rr].v;
#pragma unroll
    for (int i = 0; i < 8; ++i) {
      PairH p01, p23;
      p01.h[0] = rh[0].h[i]; p01.h[1] = rh[1].h[i];
      p23.h[0] = rh[2].h[i]; p23.h[1] = rh[3].h[i];
      *(unsigned*)&Kt[(e0 + i) * SCHUNK + 4 * sp]     = p01.u;
      *(unsigned*)&Kt[(e0 + i) * SCHUNK + 4 * sp + 2] = p23.u;
    }
    __syncthreads();

    // -------- scores transposed (already log2-scaled): 4 M-tiles of S^T
    v8f c[4];
#pragma unroll
    for (int mt = 0; mt < 4; ++mt) {
      v8f cz = {};
      v16h ak0 = frag_a_lds(&Krm[(mt * 16) * E_],      E_, lm, hf);
      v16h ak1 = frag_a_lds(&Krm[(mt * 16) * E_ + 32], E_, lm, hf);
      cz = __builtin_amdgcn_wmma_f32_16x16x32_f16(false, ak0, false, bq0,
                                                  (short)0, cz, false, false);
      cz = __builtin_amdgcn_wmma_f32_16x16x32_f16(false, ak1, false, bq1,
                                                  (short)0, cz, false, false);
      c[mt] = cz;
    }

    // -------- online softmax: in-lane max over 32 regs + one half-exchange
    float mx = c[0][0];
#pragma unroll
    for (int mt = 0; mt < 4; ++mt)
#pragma unroll
      for (int r = 0; r < 8; ++r) mx = fmaxf(mx, c[mt][r]);
    mx = fmaxf(mx, __shfl_xor(mx, 16, 32));

    float mnew = fmaxf(mrun, mx);
    float corr = exp2f(mrun - mnew);

    // exp2 + pack + store P^T as [l][s] pairs, accumulating the row sum
    _Float16* Pm = Pw[wave];
    float ps = 0.0f;
#pragma unroll
    for (int mt = 0; mt < 4; ++mt)
#pragma unroll
      for (int r = 0; r < 8; r += 2) {
        float p0 = exp2f(c[mt][r]     - mnew);
        float p1 = exp2f(c[mt][r + 1] - mnew);
        ps += p0 + p1;
        PairH pk;
        pk.h[0] = (_Float16)p0; pk.h[1] = (_Float16)p1;
        *(unsigned*)&Pm[lm * SCHUNK + mt * 16 + 8 * hf + r] = pk.u;
      }
    ps += __shfl_xor(ps, 16, 32);

    lrun = lrun * corr + ps;
    mrun = mnew;
#pragma unroll
    for (int t = 0; t < 4; ++t)
#pragma unroll
      for (int r = 0; r < 8; ++r) acc[t][r] *= corr;

    // -------- V1^T tiles: A = K^T (d x s), B = P^T (s x l); k split in two
    v16h bp0 = frag_b_lds(Pm,      SCHUNK, lm, hf);
    v16h bp1 = frag_b_lds(Pm + 32, SCHUNK, lm, hf);
#pragma unroll
    for (int t = 0; t < 4; ++t) {
      v16h a0 = frag_a_lds(&Kt[(t * 16) * SCHUNK],      SCHUNK, lm, hf);
      v16h a1 = frag_a_lds(&Kt[(t * 16) * SCHUNK + 32], SCHUNK, lm, hf);
      acc[t] = __builtin_amdgcn_wmma_f32_16x16x32_f16(false, a0, false, bp0,
                                                      (short)0, acc[t], false, false);
      acc[t] = __builtin_amdgcn_wmma_f32_16x16x32_f16(false, a1, false, bp1,
                                                      (short)0, acc[t], false, false);
    }
  }

  // -------- finalize: lane owns row l = m0+lm; write as 8 x float4
  float inv = 1.0f / lrun;
  float* orow = Out + (((size_t)b * L_ + (m0 + lm)) * H_ + h) * E_ + 8 * hf;
#pragma unroll
  for (int t = 0; t < 4; ++t) {
    float4 w0 = make_float4(acc[t][0] * inv, acc[t][1] * inv,
                            acc[t][2] * inv, acc[t][3] * inv);
    float4 w1 = make_float4(acc[t][4] * inv, acc[t][5] * inv,
                            acc[t][6] * inv, acc[t][7] * inv);
    *(float4*)(orow + t * 16)     = w0;
    *(float4*)(orow + t * 16 + 4) = w1;
  }
}

extern "C" void kernel_launch(void* const* d_in, const int* in_sizes, int n_in,
                              void* d_out, int out_size, void* d_ws, size_t ws_size,
                              hipStream_t stream) {
  (void)in_sizes; (void)n_in; (void)out_size; (void)d_ws; (void)ws_size;
  const float* q = (const float*)d_in[0];
  const float* k = (const float*)d_in[1];
  float* out = (float*)d_out;
  const size_t n1 = (size_t)B_ * L_ * H_ * E_;

  dim3 grid(L_ / BM, B_ * H_);
  // V1 = softmax(Q K^T / sqrt(E)) @ K
  attn_pass<<<grid, NT, 0, stream>>>(q, k, out);
  // V2 = softmax(K Q^T / sqrt(E)) @ Q  (column softmax of scores, aggregating Q)
  attn_pass<<<grid, NT, 0, stream>>>(k, q, out + n1);
}
